// MyModel_25769803776033
// MI455X (gfx1250) — matrix operand order repaired and checked by reading, not compiled, not run on hardware
//
#include <hip/hip_runtime.h>
#include <hip/hip_bf16.h>
#include <stdint.h>

// ---------------------------------------------------------------------------
// CGConv GNN forward for MI455X (gfx1250).
// Dominant cost: 6x GEMM [E,297]x[297,128] -> v_wmma_f32_16x16x32_bf16.
// Edge kernel: 32-edge x 128-col tile per block, 8 waves, 4 WMMAs per k-step
// per wave (2 M-subtiles x {gate, softplus} matrices), 40 WMMAs per wave.
// ---------------------------------------------------------------------------

typedef __bf16  v16bf __attribute__((ext_vector_type(16)));
typedef float   v8f   __attribute__((ext_vector_type(8)));

#define AF   92
#define ED   41
#define DIM  128
#define HDIM 256
#define NG   64
#define KPAD 320          // 297 padded up to multiple of 32
#define ZP   328          // LDS row stride in elements (multiple of 8)
#define KB   10           // K blocks of 32
#define MT   32           // edges (M) per block

__device__ __forceinline__ unsigned short f2bf(float f) {
  union { float f; unsigned int u; } x; x.f = f;
  unsigned int u = x.u;
  unsigned int r = u + 0x7FFFu + ((u >> 16) & 1u);   // round-to-nearest-even
  return (unsigned short)(r >> 16);
}

__device__ __forceinline__ float sigmoidf_(float x) {
  return 1.0f / (1.0f + __expf(-x));
}
__device__ __forceinline__ float softplusf_(float x) {
  return (x > 20.0f) ? x : log1pf(__expf(x));
}

struct BFrag { uint4 a, b; };   // 32 bytes == v16bf

// ---------------------------------------------------------------------------
// Pack Wf/Ws (fp32 [L,297,128]) into bf16 WMMA-B fragments:
// layout [l][mat][kb][ntile][lane][8 dwords]; lane holds col n = ntile*16+lane%16,
// K pair (2*d, 2*d+1) within half selected by lane>=16 (K offset +16).
// ---------------------------------------------------------------------------
__global__ void pack_weights_kernel(const float* __restrict__ Wf,
                                    const float* __restrict__ Ws,
                                    unsigned int* __restrict__ packed,
                                    int total) {
  int idx = blockIdx.x * 256 + threadIdx.x;
  if (idx >= total) return;
  int t = idx;
  int d    = t % 8;  t /= 8;
  int lane = t % 32; t /= 32;
  int nt   = t % 8;  t /= 8;
  int kb   = t % 10; t /= 10;
  int mat  = t % 2;  t /= 2;
  int l    = t;
  int k0 = kb * 32 + ((lane >> 4) * 16) + 2 * d;
  int n  = nt * 16 + (lane & 15);
  const float* W = mat ? Ws : Wf;
  size_t base = (size_t)l * 297 * DIM;
  float lo = (k0     < 297) ? W[base + (size_t)k0       * DIM + n] : 0.0f;
  float hi = (k0 + 1 < 297) ? W[base + (size_t)(k0 + 1) * DIM + n] : 0.0f;
  packed[idx] = ((unsigned int)f2bf(hi) << 16) | (unsigned int)f2bf(lo);
}

// ---------------------------------------------------------------------------
// Embedding: h = x @ emb_W + emb_b ; also emit bf16 copy of h.
// ---------------------------------------------------------------------------
__global__ __launch_bounds__(128) void embed_kernel(
    const float* __restrict__ x, const float* __restrict__ emb_W,
    const float* __restrict__ emb_b, float* __restrict__ h,
    unsigned short* __restrict__ hbf, int N) {
  __shared__ float xrow[AF];
  int n = blockIdx.x;
  if (n >= N) return;
  int d = threadIdx.x;
  if (d < AF) xrow[d] = x[(size_t)n * AF + d];
  __syncthreads();
  float acc = emb_b[d];
  #pragma unroll 4
  for (int k = 0; k < AF; ++k) acc = fmaf(xrow[k], emb_W[(size_t)k * DIM + d], acc);
  size_t o = (size_t)n * DIM + d;
  h[o]   = acc;
  hbf[o] = f2bf(acc);
}

// ---------------------------------------------------------------------------
// Edge message + scatter-add. Block = 32 edges x 128 cols, 8 waves.
// Wave w owns output columns [16w, 16w+16) for BOTH gate (Wf) and src (Ws)
// GEMMs and both 16-row M-subtiles; A fragments shared through LDS.
// ---------------------------------------------------------------------------
__global__ __launch_bounds__(256, 1) void edge_msg_kernel(
    const unsigned short* __restrict__ hbf,
    const float* __restrict__ edge_attr,
    const int* __restrict__ ei,           // [2,E] row-major: src row 0, dst row 1
    const unsigned int* __restrict__ packF,
    const unsigned int* __restrict__ packS,
    const float* __restrict__ biasF,      // bf[l], 128
    const float* __restrict__ biasS,      // bs[l], 128
    float* __restrict__ agg, int E) {
  __shared__ __attribute__((aligned(16))) unsigned short zt[MT * ZP];
  __shared__ int dstS[MT];
  __shared__ int srcS[MT];

  const int tid = threadIdx.x;
  const int e0  = blockIdx.x * MT;
  const int* srcI = ei;
  const int* dstI = ei + E;

  if (tid < MT) {
    int e = e0 + tid;
    dstS[tid] = (e < E) ? dstI[e] : 0;
  } else if (tid < 2 * MT) {
    int r = tid - MT;
    int e = e0 + r;
    srcS[r] = (e < E) ? srcI[e] : 0;
  }
  __syncthreads();

  // z tile fill, vectorized: [h[dst](128) | h[src](128)] as 16B chunks.
  // 32 rows x 32 chunks = 1024 chunks; 4 per thread.
  #pragma unroll
  for (int c = tid; c < MT * 32; c += 256) {
    int r  = c >> 5;
    int cc = (c & 31) << 3;            // bf16 column, 0..248
    int e  = e0 + r;
    uint4 v = make_uint4(0u, 0u, 0u, 0u);
    if (e < E) {
      int node = (cc < DIM) ? dstS[r] : srcS[r];
      int col  = (cc < DIM) ? cc : (cc - DIM);
      v = *(const uint4*)(hbf + (size_t)node * DIM + col);
    }
    *(uint4*)(&zt[r * ZP + cc]) = v;
  }
  // edge_attr (41 cols) + zero pad to 320: cols 256..319, scalar u16.
  #pragma unroll
  for (int idx = tid; idx < MT * 64; idx += 256) {
    int r = idx >> 6;
    int c = 256 + (idx & 63);
    int e = e0 + r;
    unsigned short v = 0;
    if (e < E && c < 2 * DIM + ED)
      v = f2bf(edge_attr[(size_t)e * ED + (c - 2 * DIM)]);
    zt[r * ZP + c] = v;
  }
  __syncthreads();

  const int wave  = tid >> 5;
  const int lane  = tid & 31;
  const int mA    = lane & 15;            // A-fragment row within subtile
  const int khalf = (lane >> 4) * 16;     // A-fragment K sub-half
  const int ng    = wave * 16 + (lane & 15);  // global output column

  v8f accF0, accF1, accS0, accS1;
  {
    float bFv = biasF[ng], bSv = biasS[ng];
    #pragma unroll
    for (int i = 0; i < 8; ++i) {
      accF0[i] = bFv; accF1[i] = bFv;
      accS0[i] = bSv; accS1[i] = bSv;
    }
  }

  #pragma unroll
  for (int kb = 0; kb < KB; ++kb) {
    // A fragments from LDS (contiguous k order: element s -> k = khalf + s, as in packed B): rows 0-15 and rows 16-31.
    int aoff = mA * ZP + kb * 32 + khalf;
    BFrag af0, af1;
    af0.a = *(const uint4*)(&zt[aoff]);
    af0.b = *(const uint4*)(&zt[aoff + 8]);
    af1.a = *(const uint4*)(&zt[aoff + 16 * ZP]);
    af1.b = *(const uint4*)(&zt[aoff + 16 * ZP + 8]);
    v16bf av0 = __builtin_bit_cast(v16bf, af0);
    v16bf av1 = __builtin_bit_cast(v16bf, af1);

    // B fragments: 8 contiguous dwords per lane from the pre-packed buffer.
    size_t fo = ((size_t)(kb * 8 + wave) * 32 + lane) * 8;
    BFrag bfF, bfS;
    bfF.a = *(const uint4*)(packF + fo);
    bfF.b = *(const uint4*)(packF + fo + 4);
    bfS.a = *(const uint4*)(packS + fo);
    bfS.b = *(const uint4*)(packS + fo + 4);
    v16bf bvF = __builtin_bit_cast(v16bf, bfF);
    v16bf bvS = __builtin_bit_cast(v16bf, bfS);

    accF0 = __builtin_amdgcn_wmma_f32_16x16x32_bf16(false, av0, false, bvF,
                                                    (short)0, accF0, false, false);
    accS0 = __builtin_amdgcn_wmma_f32_16x16x32_bf16(false, av0, false, bvS,
                                                    (short)0, accS0, false, false);
    accF1 = __builtin_amdgcn_wmma_f32_16x16x32_bf16(false, av1, false, bvF,
                                                    (short)0, accF1, false, false);
    accS1 = __builtin_amdgcn_wmma_f32_16x16x32_bf16(false, av1, false, bvS,
                                                    (short)0, accS1, false, false);
  }

  // Epilogue: msg = sigmoid(f) * softplus(s); scatter-add into agg[dst].
  const int m0 = (lane >> 4) * 8;   // C-layout: VGPR i -> row m0+i, col lane&15
  #pragma unroll
  for (int i = 0; i < 8; ++i) {
    int m = m0 + i;
    int e = e0 + m;
    if (e < E) {
      float msg = sigmoidf_(accF0[i]) * softplusf_(accS0[i]);
      atomicAdd(&agg[(size_t)dstS[m] * DIM + ng], msg);
    }
    int e2 = e + 16;
    if (e2 < E) {
      float msg = sigmoidf_(accF1[i]) * softplusf_(accS1[i]);
      atomicAdd(&agg[(size_t)dstS[m + 16] * DIM + ng], msg);
    }
  }
}

// ---------------------------------------------------------------------------
// BatchNorm(eval) + residual; refresh fp32 h and bf16 mirror.
// ---------------------------------------------------------------------------
__global__ __launch_bounds__(256) void bn_res_kernel(
    float* __restrict__ h, unsigned short* __restrict__ hbf,
    const float* __restrict__ agg,
    const float* __restrict__ gamma, const float* __restrict__ beta,
    const float* __restrict__ mean, const float* __restrict__ var, int total) {
  int i = blockIdx.x * 256 + threadIdx.x;
  if (i >= total) return;
  int d = i & (DIM - 1);
  float a = agg[i];
  float v = (a - mean[d]) * rsqrtf(var[d] + 1e-5f) * gamma[d] + beta[d];
  float hn = h[i] + v;
  h[i]   = hn;
  hbf[i] = f2bf(hn);
}

// ---------------------------------------------------------------------------
// Mean-pool per graph (atomic accumulate).
// ---------------------------------------------------------------------------
__global__ __launch_bounds__(256) void pool_kernel(
    const float* __restrict__ h, const int* __restrict__ batch,
    float* __restrict__ pooled, float* __restrict__ counts, int total) {
  int i = blockIdx.x * 256 + threadIdx.x;
  if (i >= total) return;
  int n = i >> 7, d = i & (DIM - 1);
  int b = batch[n];
  atomicAdd(&pooled[(size_t)b * DIM + d], h[i]);
  if (d == 0) atomicAdd(&counts[b], 1.0f);
}

__global__ __launch_bounds__(256) void g1_kernel(
    float* __restrict__ pooled, const float* __restrict__ counts) {
  int i = blockIdx.x * 256 + threadIdx.x;
  if (i >= NG * DIM) return;
  float c = fmaxf(counts[i >> 7], 1.0f);
  pooled[i] = softplusf_(pooled[i] / c);
}

__global__ __launch_bounds__(256) void fc_kernel(
    const float* __restrict__ g1, const float* __restrict__ fcW,
    const float* __restrict__ fcb, float* __restrict__ g2) {
  int i = blockIdx.x * 256 + threadIdx.x;
  if (i >= NG * HDIM) return;
  int b = i >> 8, hh = i & (HDIM - 1);
  float acc = fcb[hh];
  #pragma unroll 4
  for (int d = 0; d < DIM; ++d)
    acc = fmaf(g1[(size_t)b * DIM + d], fcW[(size_t)d * HDIM + hh], acc);
  g2[i] = softplusf_(acc);
}

__global__ __launch_bounds__(64) void out_kernel(
    const float* __restrict__ g2, const float* __restrict__ outW,
    const float* __restrict__ outb, float* __restrict__ out) {
  int b = blockIdx.x * 64 + threadIdx.x;
  if (b >= NG) return;
  float acc = outb[0];
  #pragma unroll 4
  for (int hh = 0; hh < HDIM; ++hh)
    acc = fmaf(g2[(size_t)b * HDIM + hh], outW[hh], acc);
  out[b] = acc;
}

// ---------------------------------------------------------------------------
extern "C" void kernel_launch(void* const* d_in, const int* in_sizes, int n_in,
                              void* d_out, int out_size, void* d_ws, size_t ws_size,
                              hipStream_t stream) {
  const float* x         = (const float*)d_in[0];
  const int*   ei        = (const int*)  d_in[1];
  const float* edge_attr = (const float*)d_in[2];
  const int*   batch     = (const int*)  d_in[3];
  const float* emb_W     = (const float*)d_in[4];
  const float* emb_b     = (const float*)d_in[5];
  const float* Wf        = (const float*)d_in[6];
  const float* bf_       = (const float*)d_in[7];
  const float* Ws        = (const float*)d_in[8];
  const float* bs_       = (const float*)d_in[9];
  const float* gamma     = (const float*)d_in[10];
  const float* beta      = (const float*)d_in[11];
  const float* mean      = (const float*)d_in[12];
  const float* var       = (const float*)d_in[13];
  const float* fcW       = (const float*)d_in[14];
  const float* fcb       = (const float*)d_in[15];
  const float* outW      = (const float*)d_in[16];
  const float* outb      = (const float*)d_in[17];

  const int N = in_sizes[0] / AF;
  const int E = in_sizes[1] / 2;
  const int L = 3;

  // Workspace carve-up (256B aligned)
  char* ws = (char*)d_ws;
  size_t off = 0;
  auto carve = [&](size_t bytes) {
    char* p = ws + off;
    off = (off + bytes + 255) & ~(size_t)255;
    return p;
  };
  const size_t matDwords   = (size_t)KB * 8 * 32 * 8;          // 20480 per matrix
  const size_t packTotal   = (size_t)L * 2 * matDwords;        // 122880 dwords
  float*          h      = (float*)         carve((size_t)N * DIM * 4);
  unsigned short* hbf    = (unsigned short*)carve((size_t)N * DIM * 2);
  float*          agg    = (float*)         carve((size_t)N * DIM * 4);
  unsigned int*   packed = (unsigned int*)  carve(packTotal * 4);
  float*          pooled = (float*)         carve((size_t)NG * DIM * 4);
  float*          counts = (float*)         carve((size_t)NG * 4);
  float*          g2     = (float*)         carve((size_t)NG * HDIM * 4);

  // 1) Pack weights into bf16 WMMA-B fragments
  pack_weights_kernel<<<((int)packTotal + 255) / 256, 256, 0, stream>>>(
      Wf, Ws, packed, (int)packTotal);

  // 2) Embedding
  embed_kernel<<<N, 128, 0, stream>>>(x, emb_W, emb_b, h, hbf, N);

  // 3) CGConv layers
  for (int l = 0; l < L; ++l) {
    hipMemsetAsync(agg, 0, (size_t)N * DIM * 4, stream);
    const unsigned int* pF = packed + (size_t)l * 2 * matDwords;
    const unsigned int* pS = pF + matDwords;
    edge_msg_kernel<<<(E + MT - 1) / MT, 256, 0, stream>>>(
        hbf, edge_attr, ei, pF, pS, bf_ + l * DIM, bs_ + l * DIM, agg, E);
    bn_res_kernel<<<(N * DIM + 255) / 256, 256, 0, stream>>>(
        h, hbf, agg, gamma + l * DIM, beta + l * DIM, mean + l * DIM,
        var + l * DIM, N * DIM);
  }

  // 4) Mean pool + MLP head
  hipMemsetAsync(pooled, 0, (size_t)NG * DIM * 4, stream);
  hipMemsetAsync(counts, 0, (size_t)NG * 4, stream);
  pool_kernel<<<(N * DIM + 255) / 256, 256, 0, stream>>>(h, batch, pooled, counts, N * DIM);
  g1_kernel<<<(NG * DIM + 255) / 256, 256, 0, stream>>>(pooled, counts);
  fc_kernel<<<(NG * HDIM + 255) / 256, 256, 0, stream>>>(pooled, fcW, fcb, g2);
  out_kernel<<<1, 64, 0, stream>>>(g2, outW, outb, (float*)d_out);
}
